// MoEMemoryLayer_81844896792936
// MI455X (gfx1250) — matrix operand order repaired and checked by reading, not compiled, not run on hardware
//
#include <hip/hip_runtime.h>

// ---------------- problem constants ----------------
#define B_  4
#define S_  1024
#define D_  1024
#define H_  16
#define HD_ 64
#define E_  8
#define CAP_ 640
#define N_  (B_ * S_)      // 4096
#define D4_ (4 * D_)       // 4096

typedef _Float16 h16;
typedef __attribute__((ext_vector_type(16))) _Float16 v16h;
typedef __attribute__((ext_vector_type(8)))  _Float16 h8;
typedef __attribute__((ext_vector_type(8)))  float    v8f;

// ---- CDNA5 async global->LDS copy (ASYNCcnt-tracked, no VGPR round trip) ----
__device__ __forceinline__ unsigned lds_addr_of(const void* p)
{
    // generic pointer to LDS: low 32 bits are the workgroup-relative LDS offset
    return (unsigned)(unsigned long long)(uintptr_t)p;
}
__device__ __forceinline__ void async_copy_b128(unsigned ldsAddr, const void* gptr)
{
    asm volatile("global_load_async_to_lds_b128 %0, %1, off"
                 :: "v"(ldsAddr), "v"((unsigned long long)(uintptr_t)gptr)
                 : "memory");
}
__device__ __forceinline__ void wait_async0()
{
    asm volatile("s_wait_asynccnt 0x0" ::: "memory");
}
__device__ __forceinline__ v16h frag_from(const h16* p0, const h16* p1)
{
    const h8 lo = *(const h8*)p0;
    const h8 hi = *(const h8*)p1;
    v16h f;
    #pragma unroll
    for (int i = 0; i < 8; ++i) { f[i] = lo[i]; f[8 + i] = hi[i]; }
    return f;
}

// =====================================================================
// LayerNorm (fp32 in) -> f16 out, one block per row
// =====================================================================
__global__ __launch_bounds__(256)
void moe_ln_f16(const float* __restrict__ x, const float* __restrict__ g,
                const float* __restrict__ bta, h16* __restrict__ out, int Dd)
{
    __shared__ float rs[8], rs2[8];
    const int row = blockIdx.x;
    const float* xr = x + (long)row * Dd;
    float s = 0.f, s2 = 0.f;
    for (int i = threadIdx.x; i < Dd; i += 256) {
        float v = xr[i]; s += v; s2 += v * v;
    }
    for (int d = 16; d >= 1; d >>= 1) {
        s  += __shfl_xor(s,  d, 32);
        s2 += __shfl_xor(s2, d, 32);
    }
    if ((threadIdx.x & 31) == 0) { rs[threadIdx.x >> 5] = s; rs2[threadIdx.x >> 5] = s2; }
    __syncthreads();
    float ts = 0.f, ts2 = 0.f;
    #pragma unroll
    for (int i = 0; i < 8; ++i) { ts += rs[i]; ts2 += rs2[i]; }
    const float mu   = ts / (float)Dd;
    const float var  = ts2 / (float)Dd - mu * mu;
    const float rinv = rsqrtf(var + 1e-5f);
    for (int i = threadIdx.x; i < Dd; i += 256)
        out[(long)row * Dd + i] = (h16)((xr[i] - mu) * rinv * g[i] + bta[i]);
}

// =====================================================================
// Transpose fp32 [R,C] -> f16 [C,R]
// =====================================================================
__global__ __launch_bounds__(256)
void moe_transpose_f16(const float* __restrict__ in, h16* __restrict__ out,
                       int R, int C, long inBatch, long outBatch)
{
    __shared__ float tile[32][33];
    const int z = blockIdx.z;
    const float* ib = in + (long)z * inBatch;
    h16* ob = out + (long)z * outBatch;
    const int r0 = blockIdx.y * 32, c0 = blockIdx.x * 32;
    const int tx = threadIdx.x & 31, ty = threadIdx.x >> 5;     // 32 x 8
    #pragma unroll
    for (int i = 0; i < 32; i += 8) {
        int r = r0 + ty + i, c = c0 + tx;
        tile[ty + i][tx] = (r < R && c < C) ? ib[(long)r * C + c] : 0.f;
    }
    __syncthreads();
    #pragma unroll
    for (int i = 0; i < 32; i += 8) {
        int r = c0 + ty + i, c = r0 + tx;                        // out is C x R
        if (r < C && c < R) ob[(long)r * R + c] = (h16)tile[tx][ty + i];
    }
}

// =====================================================================
// WMMA GEMM, 128x128 tile, K-chunk 64, double-buffered async LDS staging,
// pipeline unrolled by 2 (static buffer indices). 16 WMMA per barrier pair.
// =====================================================================
__global__ __launch_bounds__(256)
void moe_gemm_wmma(const h16* __restrict__ A, int lda, long aBatchStride,
                   const int* __restrict__ aIdx, int aIdxStride,
                   const h16* __restrict__ Bt, long btBatchStride,
                   const float* __restrict__ bias, int biasBatchStride,
                   float* __restrict__ Cf, h16* __restrict__ Ch,
                   int ldc, long cBatchStride,
                   const float* __restrict__ resid,
                   int M, int Nn, int K, int relu)
{
    __shared__ alignas(16) h16 As[2][128][64];
    __shared__ alignas(16) h16 Bs[2][128][64];

    const int z = blockIdx.z;
    const h16* Ab  = A  + (long)z * aBatchStride;
    const h16* Btb = Bt + (long)z * btBatchStride;
    const int* idx = aIdx ? (aIdx + (long)z * aIdxStride) : nullptr;

    const int tid  = threadIdx.x;
    const int lane = tid & 31;
    const int wave = tid >> 5;                 // 0..7
    const int lr   = lane & 15;
    const int half = lane >> 4;                // 0/1
    const int m0   = blockIdx.y * 128;
    const int n0   = blockIdx.x * 128;

    // staging: thread owns chunk rows r0+32j (j=0..3) at column coff (16B)
    const int r0   = tid >> 3;                 // 0..31
    const int coff = (tid & 7) << 3;           // 0..56 halves
    unsigned long long aAddr[4], bAddr[4];
    #pragma unroll
    for (int j = 0; j < 4; ++j) {
        const int gr = m0 + r0 + 32 * j;
        const long ar = idx ? (long)idx[gr] : (long)gr;
        aAddr[j] = (unsigned long long)(uintptr_t)(Ab + ar * lda + coff);
        bAddr[j] = (unsigned long long)(uintptr_t)(Btb + (long)(n0 + r0 + 32 * j) * K + coff);
    }
    unsigned lA[2], lB[2];
    #pragma unroll
    for (int p = 0; p < 2; ++p) {
        lA[p] = lds_addr_of(&As[p][r0][coff]);
        lB[p] = lds_addr_of(&Bs[p][r0][coff]);
    }

    v8f acc[8] = {};

    // stage K-chunk into buffer p, advance addresses by 64 halves
    auto stage = [&](int p) {
        #pragma unroll
        for (int j = 0; j < 4; ++j) {
            async_copy_b128(lA[p] + 4096u * j, (const void*)(uintptr_t)aAddr[j]);
            async_copy_b128(lB[p] + 4096u * j, (const void*)(uintptr_t)bAddr[j]);
            aAddr[j] += 128; bAddr[j] += 128;
        }
    };
    // 16 WMMAs on buffer p (2 d-chunks x 8 n-tiles, batched frag loads)
    auto compute = [&](int p) {
        #pragma unroll
        for (int dc = 0; dc < 2; ++dc) {
            const v16h af = frag_from(&As[p][wave * 16 + lr][dc * 32 + half * 16],
                                      &As[p][wave * 16 + lr][dc * 32 + half * 16 + 8]);
            #pragma unroll
            for (int g = 0; g < 2; ++g) {       // two batches of 4 n-tiles
                v16h bf[4];
                #pragma unroll
                for (int t = 0; t < 4; ++t)
                    bf[t] = frag_from(&Bs[p][(g * 4 + t) * 16 + lr][dc * 32 + half * 16],
                                      &Bs[p][(g * 4 + t) * 16 + lr][dc * 32 + half * 16 + 8]);
                #pragma unroll
                for (int t = 0; t < 4; ++t)
                    acc[g * 4 + t] = __builtin_amdgcn_wmma_f32_16x16x32_f16(
                        false, af, false, bf[t], (short)0, acc[g * 4 + t], false, false);
            }
        }
    };

    stage(0);
    int k0 = 0;
    while (true) {
        wait_async0();
        __syncthreads();                       // buffer 0 ready
        if (k0 + 64 < K) stage(1);
        compute(0);
        __syncthreads();
        k0 += 64;
        if (k0 >= K) break;

        wait_async0();
        __syncthreads();                       // buffer 1 ready
        if (k0 + 64 < K) stage(0);
        compute(1);
        __syncthreads();
        k0 += 64;
        if (k0 >= K) break;
    }

    // ---- epilogue: C layout (m = vgpr + 8*half, n = lr) ----
    float* Cb  = Cf ? (Cf + (long)z * cBatchStride) : nullptr;
    h16*   Chb = Ch ? (Ch + (long)z * cBatchStride) : nullptr;
    const int wm = m0 + wave * 16;
    #pragma unroll
    for (int t = 0; t < 8; ++t) {
        const int n  = n0 + t * 16 + lr;
        const float bv = bias ? bias[(long)z * biasBatchStride + n] : 0.f;
        #pragma unroll
        for (int i = 0; i < 8; ++i) {
            const int m = wm + i + half * 8;
            float vv = acc[t][i] + bv;
            if (relu && vv < 0.f) vv = 0.f;
            if (resid) vv += resid[(long)m * ldc + n];
            const long off = (long)m * ldc + n;
            if (Chb) Chb[off] = (h16)vv;
            else     Cb[off]  = vv;
        }
    }
}

// =====================================================================
// Causal flash attention, WMMA. grid (S/64, H, B), block 128 (4 waves).
// Async K staging; V transposed through VGPRs. Online softmax via shfl.
// Fragment loads batched so WMMAs issue back-to-back.
// =====================================================================
__global__ __launch_bounds__(128)
void moe_attn_wmma(const h16* __restrict__ q, const h16* __restrict__ k,
                   const h16* __restrict__ v, h16* __restrict__ o)
{
    __shared__ alignas(16) h16 Ks[32][64];        // [kv][d]
    __shared__ alignas(16) h16 Vt[64][32];        // [d][kv]
    __shared__ alignas(16) h16 Ps[4][16][32];     // per-wave P scratch

    const int b    = blockIdx.z, h = blockIdx.y;
    const int qb   = blockIdx.x * 64;
    const int tid  = threadIdx.x;
    const int lane = tid & 31, wave = tid >> 5;
    const int lr   = lane & 15, half = lane >> 4;
    const int qrow_base = qb + wave * 16;
    const long base = ((long)b * S_) * D_ + (long)h * HD_;

    // Q fragments for d-chunks [0,32) and [32,64), contiguous k order (k = 16*half + s)
    v16h qf[2];
    #pragma unroll
    for (int c = 0; c < 2; ++c) {
        const h16* qp = q + base + (long)(qrow_base + lr) * D_ + c * 32 + half * 16;
        qf[c] = frag_from(qp, qp + 8);
    }

    // hoisted staging addresses: thread owns K/V rows r0, r0+16 at col coff
    const int r0   = tid >> 3;                 // 0..15
    const int coff = (tid & 7) << 3;           // 0..56 halves
    const h16* kRow0 = k + base + (long)r0 * D_ + coff;
    const h16* vRow0 = v + base + (long)r0 * D_ + coff;
    const unsigned lK0 = lds_addr_of(&Ks[r0][coff]);
    const unsigned lK1 = lds_addr_of(&Ks[r0 + 16][coff]);

    float rmax[8], rsum[8];
    v8f oacc[4] = {};
    #pragma unroll
    for (int i = 0; i < 8; ++i) { rmax[i] = -3e30f; rsum[i] = 0.f; }

    const float scale = 0.125f;                    // 1/sqrt(64)
    const int kend = qb + 64;                      // causal block bound

    for (int kc = 0; kc < kend; kc += 32) {
        // async K rows, V via VGPR transpose
        const long koff = (long)kc * D_;
        async_copy_b128(lK0, kRow0 + koff);
        async_copy_b128(lK1, kRow0 + koff + (long)16 * D_);
        const h8 v0 = *(const h8*)(vRow0 + koff);
        const h8 v1 = *(const h8*)(vRow0 + koff + (long)16 * D_);
        #pragma unroll
        for (int i = 0; i < 8; ++i) {
            Vt[coff + i][r0]      = v0[i];
            Vt[coff + i][r0 + 16] = v1[i];
        }
        wait_async0();
        __syncthreads();

        // scores: batch all 4 B fragments, then 4 back-to-back WMMAs
        v16h kf[2][2];
        #pragma unroll
        for (int t = 0; t < 2; ++t)
            #pragma unroll
            for (int dc = 0; dc < 2; ++dc)
                kf[t][dc] = frag_from(&Ks[t * 16 + lr][dc * 32 + half * 16],
                                      &Ks[t * 16 + lr][dc * 32 + half * 16 + 8]);
        v8f st[2] = {};
        #pragma unroll
        for (int t = 0; t < 2; ++t)
            #pragma unroll
            for (int dc = 0; dc < 2; ++dc)
                st[t] = __builtin_amdgcn_wmma_f32_16x16x32_f16(
                    false, qf[dc], false, kf[t][dc], (short)0, st[t], false, false);

        // scale + causal mask + online softmax update
        #pragma unroll
        for (int i = 0; i < 8; ++i) {
            const int qg = qrow_base + i + half * 8;
            float mx = rmax[i];
            #pragma unroll
            for (int t = 0; t < 2; ++t) {
                const int kg = kc + t * 16 + lr;
                float sv = st[t][i] * scale;
                if (kg > qg) sv = -3e30f;
                st[t][i] = sv;
                mx = fmaxf(mx, sv);
            }
            for (int d = 8; d >= 1; d >>= 1) mx = fmaxf(mx, __shfl_xor(mx, d, 32));
            const float alpha = __expf(rmax[i] - mx);
            rmax[i] = mx;
            rsum[i] *= alpha;
            #pragma unroll
            for (int t = 0; t < 4; ++t) oacc[t][i] *= alpha;
            float psum = 0.f;
            #pragma unroll
            for (int t = 0; t < 2; ++t) {
                const float p = __expf(st[t][i] - mx);
                st[t][i] = p;
                psum += p;
            }
            for (int d = 8; d >= 1; d >>= 1) psum += __shfl_xor(psum, d, 32);
            rsum[i] += psum;
        }

        // C-layout f32 -> A-layout f16 via per-wave LDS scratch
        #pragma unroll
        for (int i = 0; i < 8; ++i) {
            const int m = i + half * 8;
            Ps[wave][m][lr]      = (h16)st[0][i];
            Ps[wave][m][16 + lr] = (h16)st[1][i];
        }
        __syncthreads();

        const v16h pf = frag_from(&Ps[wave][lr][half * 16],
                                  &Ps[wave][lr][half * 16 + 8]);
        // O += P(16x32) x V(32x64): batch 4 fragments, 4 WMMAs
        v16h vf[4];
        #pragma unroll
        for (int t = 0; t < 4; ++t)
            vf[t] = frag_from(&Vt[t * 16 + lr][half * 16],
                              &Vt[t * 16 + lr][half * 16 + 8]);
        #pragma unroll
        for (int t = 0; t < 4; ++t)
            oacc[t] = __builtin_amdgcn_wmma_f32_16x16x32_f16(
                false, pf, false, vf[t], (short)0, oacc[t], false, false);
        __syncthreads();
    }

    // normalize + store f16
    #pragma unroll
    for (int i = 0; i < 8; ++i) {
        const float inv = 1.f / rsum[i];
        const long row = (long)(qrow_base + i + half * 8);
        #pragma unroll
        for (int t = 0; t < 4; ++t)
            o[base + row * D_ + t * 16 + lr] = (h16)(oacc[t][i] * inv);
    }
}

// =====================================================================
// Router stage 2: logits[n][e] = r1[n,:] . Wr2[:,e] + br2[e]
// =====================================================================
__global__ __launch_bounds__(256)
void moe_router2(const h16* __restrict__ r1, const float* __restrict__ Wr2,
                 const float* __restrict__ br2, float* __restrict__ logits)
{
    __shared__ h16 row[D4_];
    const int n = blockIdx.x;
    for (int i = threadIdx.x; i < D4_; i += 256) row[i] = r1[(long)n * D4_ + i];
    __syncthreads();
    const int e = threadIdx.x >> 5, lane = threadIdx.x & 31;
    float s = 0.f;
    for (int k2 = lane; k2 < D4_; k2 += 32)
        s += (float)row[k2] * Wr2[(long)k2 * E_ + e];
    for (int d = 16; d >= 1; d >>= 1) s += __shfl_xor(s, d, 32);
    if (lane == 0) logits[(long)n * E_ + e] = s + br2[e];
}

// =====================================================================
// Expert-choice top-CAP selection (sorted desc), one block per expert
// =====================================================================
__global__ __launch_bounds__(256)
void moe_topk(const float* __restrict__ logits, float* __restrict__ scr,
              int* __restrict__ idx)
{
    const int e = blockIdx.x;
    float* se = scr + (long)e * N_;
    for (int i = threadIdx.x; i < N_; i += 256) se[i] = logits[(long)i * E_ + e];
    __syncthreads();
    __shared__ float bv[256];
    __shared__ int   bi[256];
    for (int c = 0; c < CAP_; ++c) {
        float best = -3e38f; int bidx = 0x7fffffff;
        for (int i = threadIdx.x; i < N_; i += 256) {
            const float vv = se[i];
            if (vv > best || (vv == best && i < bidx)) { best = vv; bidx = i; }
        }
        bv[threadIdx.x] = best; bi[threadIdx.x] = bidx;
        __syncthreads();
        for (int s = 128; s >= 1; s >>= 1) {
            if (threadIdx.x < s) {
                const float v2 = bv[threadIdx.x + s]; const int i2 = bi[threadIdx.x + s];
                if (v2 > bv[threadIdx.x] ||
                    (v2 == bv[threadIdx.x] && i2 < bi[threadIdx.x])) {
                    bv[threadIdx.x] = v2; bi[threadIdx.x] = i2;
                }
            }
            __syncthreads();
        }
        if (threadIdx.x == 0) { idx[(long)e * CAP_ + c] = bi[0]; se[bi[0]] = -3e38f; }
        __syncthreads();
    }
}

// =====================================================================
// zero / scatter / combine / losses
// =====================================================================
__global__ __launch_bounds__(256)
void moe_zero(float* __restrict__ p, long n)
{
    for (long i = (long)blockIdx.x * 256 + threadIdx.x; i < n; i += (long)gridDim.x * 256)
        p[i] = 0.f;
}

__global__ __launch_bounds__(256)
void moe_scatter(const float* __restrict__ eout, const int* __restrict__ idx,
                 float* __restrict__ comb, float* __restrict__ counts)
{
    const int ec  = blockIdx.x;          // e*CAP + c
    const int tok = idx[ec];
    const float* src = eout + (long)ec * D_;
    float* dst = comb + (long)tok * D_;
    for (int i = threadIdx.x; i < D_; i += 256) atomicAdd(&dst[i], src[i]);
    if (threadIdx.x == 0) atomicAdd(&counts[tok], 1.f);
}

__global__ __launch_bounds__(256)
void moe_final(const float* __restrict__ h, const float* __restrict__ comb,
               const float* __restrict__ counts, float* __restrict__ out)
{
    const int tok = blockIdx.x;
    const float inv = 1.f / fmaxf(counts[tok], 1.f);
    for (int i = threadIdx.x; i < D_; i += 256)
        out[(long)tok * D_ + i] = h[(long)tok * D_ + i] + comb[(long)tok * D_ + i] * inv;
}

__global__ __launch_bounds__(256)
void moe_losses(const float* __restrict__ logits, float* __restrict__ out, long outOff)
{
    float imp[E_] = {};
    for (int n = threadIdx.x; n < N_; n += 256) {
        float l[E_], mx = -3e38f;
        #pragma unroll
        for (int e = 0; e < E_; ++e) { l[e] = logits[(long)n * E_ + e]; mx = fmaxf(mx, l[e]); }
        float s = 0.f;
        #pragma unroll
        for (int e = 0; e < E_; ++e) { l[e] = __expf(l[e] - mx); s += l[e]; }
        const float inv = 1.f / s;
        #pragma unroll
        for (int e = 0; e < E_; ++e) imp[e] += l[e] * inv;
    }
    #pragma unroll
    for (int e = 0; e < E_; ++e)
        for (int d = 16; d >= 1; d >>= 1) imp[e] += __shfl_xor(imp[e], d, 32);
    __shared__ float wsum[8][E_];
    const int lane = threadIdx.x & 31, w = threadIdx.x >> 5;
    if (lane == 0)
        for (int e = 0; e < E_; ++e) wsum[w][e] = imp[e];
    __syncthreads();
    if (threadIdx.x == 0) {
        float tot[E_], mean = 0.f;
        for (int e = 0; e < E_; ++e) {
            tot[e] = 0.f;
            for (int ww = 0; ww < 8; ++ww) tot[e] += wsum[ww][e];
            mean += tot[e];
        }
        mean /= (float)E_;
        float var = 0.f;
        for (int e = 0; e < E_; ++e) { const float d2 = tot[e] - mean; var += d2 * d2; }
        var /= (float)(E_ - 1);
        out[outOff]     = 0.f;                               // load_loss (constant load)
        out[outOff + 1] = var / (mean * mean + 1e-6f);       // importance_loss
    }
}

// =====================================================================
// launcher
// =====================================================================
static inline void* ws_take(void* base, size_t& off, size_t bytes)
{
    off = (off + 255) & ~(size_t)255;
    void* p = (char*)base + off;
    off += bytes;
    return p;
}

extern "C" void kernel_launch(void* const* d_in, const int* in_sizes, int n_in,
                              void* d_out, int out_size, void* d_ws, size_t ws_size,
                              hipStream_t stream)
{
    const float* x   = (const float*)d_in[0];
    const float* g1  = (const float*)d_in[1];
    const float* b1  = (const float*)d_in[2];
    const float* g2  = (const float*)d_in[3];
    const float* b2  = (const float*)d_in[4];
    const float* Wq  = (const float*)d_in[5];
    const float* Wk  = (const float*)d_in[6];
    const float* Wv  = (const float*)d_in[7];
    const float* Wo  = (const float*)d_in[8];
    const float* Wr1 = (const float*)d_in[9];
    const float* br1 = (const float*)d_in[10];
    const float* Wr2 = (const float*)d_in[11];
    const float* br2 = (const float*)d_in[12];
    const float* We1 = (const float*)d_in[13];
    const float* be1 = (const float*)d_in[14];
    const float* We2 = (const float*)d_in[15];
    const float* be2 = (const float*)d_in[16];
    float* out = (float*)d_out;

    size_t off = 0;
    h16*  xn    = (h16*) ws_take(d_ws, off, (size_t)N_ * D_ * 2);
    h16*  WqT   = (h16*) ws_take(d_ws, off, (size_t)D_ * D_ * 2);
    h16*  WkT   = (h16*) ws_take(d_ws, off, (size_t)D_ * D_ * 2);
    h16*  WvT   = (h16*) ws_take(d_ws, off, (size_t)D_ * D_ * 2);
    h16*  WoT   = (h16*) ws_take(d_ws, off, (size_t)D_ * D_ * 2);
    h16*  qb    = (h16*) ws_take(d_ws, off, (size_t)N_ * D_ * 2);
    h16*  kb    = (h16*) ws_take(d_ws, off, (size_t)N_ * D_ * 2);
    h16*  vb    = (h16*) ws_take(d_ws, off, (size_t)N_ * D_ * 2);
    h16*  attno = (h16*) ws_take(d_ws, off, (size_t)N_ * D_ * 2);
    float* hbuf = (float*)ws_take(d_ws, off, (size_t)N_ * D_ * 4);
    h16*  x2    = (h16*) ws_take(d_ws, off, (size_t)N_ * D_ * 2);
    h16*  Wr1T  = (h16*) ws_take(d_ws, off, (size_t)D_ * D4_ * 2);
    h16*  r1    = (h16*) ws_take(d_ws, off, (size_t)N_ * D4_ * 2);
    float* logits = (float*)ws_take(d_ws, off, (size_t)N_ * E_ * 4);
    float* scr    = (float*)ws_take(d_ws, off, (size_t)E_ * N_ * 4);
    int*   tidx   = (int*)  ws_take(d_ws, off, (size_t)E_ * CAP_ * 4);
    h16*  We1T  = (h16*) ws_take(d_ws, off, (size_t)E_ * D_ * D4_ * 2);
    h16*  We2T  = (h16*) ws_take(d_ws, off, (size_t)E_ * D_ * D4_ * 2);
    h16*  hid   = (h16*) ws_take(d_ws, off, (size_t)E_ * CAP_ * D4_ * 2);
    float* eout = (float*)ws_take(d_ws, off, (size_t)E_ * CAP_ * D_ * 4);
    float* comb = (float*)ws_take(d_ws, off, (size_t)N_ * D_ * 4);
    float* cnts = (float*)ws_take(d_ws, off, (size_t)N_ * 4);

    // ---- attention ----
    moe_ln_f16<<<N_, 256, 0, stream>>>(x, g1, b1, xn, D_);

    dim3 tgrid(D_ / 32, D_ / 32, 1);
    moe_transpose_f16<<<tgrid, 256, 0, stream>>>(Wq, WqT, D_, D_, 0, 0);
    moe_transpose_f16<<<tgrid, 256, 0, stream>>>(Wk, WkT, D_, D_, 0, 0);
    moe_transpose_f16<<<tgrid, 256, 0, stream>>>(Wv, WvT, D_, D_, 0, 0);
    moe_transpose_f16<<<tgrid, 256, 0, stream>>>(Wo, WoT, D_, D_, 0, 0);

    dim3 gproj(D_ / 128, N_ / 128, 1);
    moe_gemm_wmma<<<gproj, 256, 0, stream>>>(xn, D_, 0, nullptr, 0, WqT, 0,
        nullptr, 0, nullptr, qb, D_, 0, nullptr, N_, D_, D_, 0);
    moe_gemm_wmma<<<gproj, 256, 0, stream>>>(xn, D_, 0, nullptr, 0, WkT, 0,
        nullptr, 0, nullptr, kb, D_, 0, nullptr, N_, D_, D_, 0);
    moe_gemm_wmma<<<gproj, 256, 0, stream>>>(xn, D_, 0, nullptr, 0, WvT, 0,
        nullptr, 0, nullptr, vb, D_, 0, nullptr, N_, D_, D_, 0);

    dim3 agrid(S_ / 64, H_, B_);
    moe_attn_wmma<<<agrid, 128, 0, stream>>>(qb, kb, vb, attno);

    // h = x + attno @ Wo   (f32 out + residual)
    moe_gemm_wmma<<<gproj, 256, 0, stream>>>(attno, D_, 0, nullptr, 0, WoT, 0,
        nullptr, 0, hbuf, nullptr, D_, 0, x, N_, D_, D_, 0);

    // ---- MoE ----
    moe_ln_f16<<<N_, 256, 0, stream>>>(hbuf, g2, b2, x2, D_);

    dim3 tr1(D4_ / 32, D_ / 32, 1);
    moe_transpose_f16<<<tr1, 256, 0, stream>>>(Wr1, Wr1T, D_, D4_, 0, 0);

    dim3 gr1(D4_ / 128, N_ / 128, 1);
    moe_gemm_wmma<<<gr1, 256, 0, stream>>>(x2, D_, 0, nullptr, 0, Wr1T, 0,
        br1, 0, nullptr, r1, D4_, 0, nullptr, N_, D4_, D_, 1);

    moe_router2<<<N_, 256, 0, stream>>>(r1, Wr2, br2, logits);
    moe_topk<<<E_, 256, 0, stream>>>(logits, scr, tidx);

    dim3 te1(D4_ / 32, D_ / 32, E_);
    moe_transpose_f16<<<te1, 256, 0, stream>>>(We1, We1T, D_, D4_,
        (long)D_ * D4_, (long)D4_ * D_);
    dim3 te2(D_ / 32, D4_ / 32, E_);
    moe_transpose_f16<<<te2, 256, 0, stream>>>(We2, We2T, D4_, D_,
        (long)D4_ * D_, (long)D_ * D4_);

    // hid[e] = relu(x2[idx[e]] @ We1[e] + be1[e])   (gathered A rows)
    dim3 ge1(D4_ / 128, CAP_ / 128, E_);
    moe_gemm_wmma<<<ge1, 256, 0, stream>>>(x2, D_, 0, tidx, CAP_,
        We1T, (long)D4_ * D_, be1, D4_, nullptr, hid, D4_, (long)CAP_ * D4_,
        nullptr, CAP_, D4_, D_, 1);

    // eout[e] = hid[e] @ We2[e] + be2[e]
    dim3 ge2(D_ / 128, CAP_ / 128, E_);
    moe_gemm_wmma<<<ge2, 256, 0, stream>>>(hid, D4_, (long)CAP_ * D4_, nullptr, 0,
        We2T, (long)D_ * D4_, be2, D_, eout, nullptr, D_, (long)CAP_ * D_,
        nullptr, CAP_, D_, D4_, 0);

    // combine
    moe_zero<<<2048, 256, 0, stream>>>(comb, (long)N_ * D_);
    moe_zero<<<32, 256, 0, stream>>>(cnts, (long)N_);
    moe_scatter<<<E_ * CAP_, 256, 0, stream>>>(eout, tidx, comb, cnts);
    moe_final<<<N_, 256, 0, stream>>>(hbuf, comb, cnts, out);

    // aux losses
    moe_losses<<<1, 256, 0, stream>>>(logits, out, (long)N_ * D_);
}
